// BiDirectionalFusionModule_61375082660503
// MI455X (gfx1250) — matrix operand order repaired and checked by reading, not compiled, not run on hardware
//
#include <hip/hip_runtime.h>
#include <hip/hip_bf16.h>
#include <math.h>

// ---------------------------------------------------------------------------
// BiDirectionalFusionModule on MI455X (gfx1250): bf16 WMMA for every matmul-
// shaped op, f32 accumulate + f32 epilogues; TDM (tensor_load_to_lds) stages
// weight/activation tiles into LDS where the tile shape allows it.
// ---------------------------------------------------------------------------

typedef __bf16 bf16;
typedef __attribute__((ext_vector_type(16))) __bf16 v16bf;
typedef __attribute__((ext_vector_type(8)))  __bf16 bf16x8;
typedef __attribute__((ext_vector_type(4)))  __bf16 bf16x4;
typedef __attribute__((ext_vector_type(8)))  float  v8f;
typedef __attribute__((ext_vector_type(4)))  float  f32x4;
typedef __attribute__((ext_vector_type(4)))  unsigned int u32x4;
typedef __attribute__((ext_vector_type(8)))  int    i32x8;
typedef __attribute__((ext_vector_type(4)))  int    i32x4;

#define NB    8
#define CCH   256
#define HH    88
#define NPIX  7744        // 88*88
#define NRED  121         // 11*11
#define NREDP 128
#define CQ    32
#define KC1   4608        // 512*9
#define KFUS  4617        // 513*9
#define KFUSP 4640        // padded to multiple of 32 (and 8) for aligned rows
#define KSR   16384       // 256*64
#define LNEPS 1e-5f

// Tiling: block 128x64x32, 8 waves (4Mx2N), wave tile 32x32 => 4 WMMAs/step.
#define BM 128
#define BN 64
#define BK 32
#define APITCH 40   // LDS pitch for As[m][k] (bf16): 64B data + 16B pad = 80B rows
#define BPITCH 68   // LDS pitch for Bs[k][n] (bf16): 128B data + 8B pad = 136B rows

#ifndef USE_TDM
#define USE_TDM 1
#endif
#if USE_TDM && defined(__gfx1250__) && __has_builtin(__builtin_amdgcn_tensor_load_to_lds)
#define HAVE_TDM 1
#else
#define HAVE_TDM 0
#endif

// ---------------------------------------------------------------------------
// WMMA core (wave32, 16x16x32 bf16 -> f32)
// ---------------------------------------------------------------------------
__device__ __forceinline__ v8f wmma_bf16(v16bf a, v16bf b, v8f c) {
#if defined(__gfx1250__)
  return __builtin_amdgcn_wmma_f32_16x16x32_bf16(false, a, false, b, (short)0, c,
                                                 false, false);
#else
  return c; // host pass: never executed
#endif
}

// LDS byte offset of a __shared__ pointer = low 32 bits of its flat address
// (LDS aperture maps addr[31:0] to the workgroup's LDS allocation).
__device__ __forceinline__ unsigned lds_off(const void* p) {
  return (unsigned)(size_t)p;
}

// TDM: load a 2D tile (tileD0 elems/row x tileD1 rows, 2-byte elements) from
// global (row stride strideElems) into LDS at ldsAddr with row padding.
// padInterval: code for DWORDs-per-row (3=16dw/64B, 4=32dw/128B);
// padAmount:   pad DWORDs after each row (encoded as amount-1), 0 = no padding.
__device__ __forceinline__ void tdm_load_2d(unsigned long long gaddr, unsigned ldsAddr,
                                            unsigned tileD0, unsigned tileD1,
                                            unsigned tensD0, unsigned tensD1,
                                            unsigned long long strideElems,
                                            unsigned padInterval, unsigned padAmount) {
#if HAVE_TDM
  u32x4 g0;
  g0[0] = 1u;                                               // count=1, user desc
  g0[1] = ldsAddr;                                          // lds_addr (bytes)
  g0[2] = (unsigned)(gaddr & 0xffffffffull);                // global_addr[31:0]
  g0[3] = (unsigned)((gaddr >> 32) & 0x01ffffffull) | (2u << 30); // [56:32]+type=2
  i32x8 g1;
  unsigned w0 = (1u << 16);                                 // data_size = 2 bytes
  if (padAmount) w0 |= (1u << 20) | (padInterval << 22) | ((padAmount - 1u) << 25);
  g1[0] = (int)w0;
  g1[1] = (int)((tensD0 & 0xffffu) << 16);                  // tensor_dim0 lo
  g1[2] = (int)((tensD0 >> 16) | ((tensD1 & 0xffffu) << 16));
  g1[3] = (int)((tensD1 >> 16) | (tileD0 << 16));           // tile_dim0
  g1[4] = (int)tileD1;                                      // tile_dim1 (tile_dim2=0)
  g1[5] = (int)(strideElems & 0xffffffffull);               // tensor_dim0_stride lo
  g1[6] = (int)((strideElems >> 32) & 0xffffull);           // stride hi (dim1 stride=0)
  g1[7] = 0;
  i32x4 z4 = {};
  i32x8 z8 = {};
  __builtin_amdgcn_tensor_load_to_lds(g0, g1, z4, z4, z8, 0);
#else
  (void)gaddr; (void)ldsAddr; (void)tileD0; (void)tileD1;
  (void)tensD0; (void)tensD1; (void)strideElems; (void)padInterval; (void)padAmount;
#endif
}

__device__ __forceinline__ void tdm_wait() {
#if HAVE_TDM
  __builtin_amdgcn_s_wait_tensorcnt(0);
#endif
}

// A fragment (16x32 bf16): lane l: m=l&15, khalf=(l>>4)*8.
// a[0..7] = K khalf..khalf+7 ; a[8..15] = K 16+khalf..16+khalf+7 (two 16B chunks)
__device__ __forceinline__ void frag_a(const bf16* As, int m0, int lane, v16bf& a) {
  const bf16* row = As + (size_t)(m0 + (lane & 15)) * APITCH + ((lane >> 4) << 3);
  bf16x8 lo = *(const bf16x8*)row;
  bf16x8 hi = *(const bf16x8*)(row + 16);
#pragma unroll
  for (int j = 0; j < 8; ++j) { a[j] = lo[j]; a[8 + j] = hi[j]; }
}

// B fragment (32x16 bf16): lane l: n=n0+(l&15), khalf=(l>>4)*8; b[0..7] = Bs[khalf+e][n], b[8..15] = Bs[16+khalf+e-8][n]
__device__ __forceinline__ void frag_b(const bf16* Bs, int n0, int lane, v16bf& b) {
  const bf16* p = Bs + (size_t)((lane >> 4) << 3) * BPITCH + n0 + (lane & 15);
#pragma unroll
  for (int e = 0; e < 16; ++e) b[e] = p[(size_t)(e < 8 ? e : e + 8) * BPITCH];
}

__device__ __forceinline__ void stage_b_row(bf16* Bs, int kk, int nseg, const bf16* tmp) {
  bf16* dst = Bs + (size_t)kk * BPITCH + nseg;   // 8B-aligned
  bf16x4 lo = {tmp[0], tmp[1], tmp[2], tmp[3]};
  bf16x4 hi = {tmp[4], tmp[5], tmp[6], tmp[7]};
  *(bf16x4*)dst = lo;
  *(bf16x4*)(dst + 4) = hi;
}

// ---------------------------------------------------------------------------
// Generic GEMM: D[b] = epilogue(A[b] * B[b]) with A bf16 row-major [M x lda].
// B modes: 0 f32 channel-major, 1 bf16 channel-major, 2/3 sr-conv gather f32/bf16.
// Epilogue: out = acc*alpha + bias[m]; store f32 or bf16. K, lda multiples of 32/8.
// ---------------------------------------------------------------------------
__global__ __launch_bounds__(256) void gemm_wmma(
    int M, int K, int N,
    const bf16* __restrict__ A, int lda, long long aStride,
    const void* __restrict__ Bsrc, int bMode, int bPitch, long long bStride,
    void* __restrict__ D, long long dStride, int dBf16,
    const float* __restrict__ bias, float alpha) {
  __shared__ bf16 As[BM * APITCH];
  __shared__ bf16 Bs[BK * BPITCH];

  const int bz   = blockIdx.z;
  const int tid  = threadIdx.x;
  const int lane = tid & 31;
  const int wid  = tid >> 5;
  const int wm   = wid >> 1;      // 0..3 -> 32-row slab
  const int wn   = wid & 1;       // 0..1 -> 32-col slab
  const int blkM = blockIdx.y * BM;
  const int blkN = blockIdx.x * BN;

  const bf16* Ab = A + (size_t)bz * (size_t)aStride;
  const bool fastB = (bMode <= 1) && (blkN + BN <= N) &&
                     ((bMode == 0) ? ((bPitch & 3) == 0) : ((bPitch & 7) == 0));
  const bool tdmA = HAVE_TDM && (blkM + BM <= M);
  const bool tdmB = HAVE_TDM && fastB && (bMode == 1);

  const int skk  = tid >> 3;          // 0..31
  const int snsg = (tid & 7) << 3;    // 0,8,..,56
  const unsigned asOff = lds_off(As);
  const unsigned bsOff = lds_off(Bs);

  v8f c00 = {}; v8f c01 = {}; v8f c10 = {}; v8f c11 = {};

  for (int k0 = 0; k0 < K; k0 += BK) {
    // ---- stage A tile [128 x 32] ----
    if (tdmA) {
      if (wid == 0) {
        unsigned long long ga =
            (unsigned long long)(size_t)(Ab + (size_t)blkM * lda + k0);
        tdm_load_2d(ga, asOff, BK, BM, (unsigned)(K - k0), (unsigned)(M - blkM),
                    (unsigned long long)lda, /*16dw rows*/3, /*pad 4dw*/4);
      }
    } else {
#pragma unroll
      for (int i = 0; i < 2; ++i) {
        int idx  = tid + i * 256;          // 0..511
        int kseg = (idx & 3) << 3;         // 0,8,16,24
        int mm   = idx >> 2;               // 0..127
        int gm   = blkM + mm;
        bf16x8 v = {};
        if (gm < M) v = *(const bf16x8*)(Ab + (size_t)gm * lda + (k0 + kseg));
        *(bf16x8*)(As + (size_t)mm * APITCH + kseg) = v;
      }
    }
    // ---- stage B tile [32 x 64] into Bs[k][n] ----
    if (tdmB) {
      if (wid == 1) {
        unsigned long long gb = (unsigned long long)(size_t)(
            (const bf16*)Bsrc + (size_t)bz * bStride + (size_t)k0 * bPitch + blkN);
        tdm_load_2d(gb, bsOff, BN, BK, (unsigned)(N - blkN), (unsigned)(K - k0),
                    (unsigned long long)bPitch, /*32dw rows*/4, /*pad 2dw*/2);
      }
    } else {
      int gk  = k0 + skk;
      int gn0 = blkN + snsg;
      bf16 tmp[8];
#pragma unroll
      for (int j = 0; j < 8; ++j) tmp[j] = (bf16)0.0f;
      if (fastB) {
        if (bMode == 0) {
          const float* p = (const float*)Bsrc + (size_t)bz * bStride +
                           (size_t)gk * bPitch + gn0;
          f32x4 u0 = *(const f32x4*)p;
          f32x4 u1 = *(const f32x4*)(p + 4);
#pragma unroll
          for (int j = 0; j < 4; ++j) { tmp[j] = (bf16)u0[j]; tmp[4 + j] = (bf16)u1[j]; }
          if (k0 + BK < K) __builtin_prefetch(p + (size_t)BK * bPitch, 0, 1);
        } else {
          const bf16* p = (const bf16*)Bsrc + (size_t)bz * bStride +
                          (size_t)gk * bPitch + gn0;
          bf16x8 w = *(const bf16x8*)p;
#pragma unroll
          for (int j = 0; j < 8; ++j) tmp[j] = w[j];
          if (k0 + BK < K) __builtin_prefetch(p + (size_t)BK * bPitch, 0, 1);
        }
      } else {
#pragma unroll
        for (int j = 0; j < 8; ++j) {
          int gn = gn0 + j;
          if (gn < N && gk < K) {
            if (bMode == 0) {
              tmp[j] = (bf16)((const float*)Bsrc)[(size_t)bz * bStride +
                                                  (size_t)gk * bPitch + gn];
            } else if (bMode == 1) {
              tmp[j] = ((const bf16*)Bsrc)[(size_t)bz * bStride +
                                           (size_t)gk * bPitch + gn];
            } else {
              int ci = gk >> 6, r = gk & 63, kh = r >> 3, kw = r & 7;
              int py = gn / 11, px = gn - py * 11;
              size_t off = (size_t)bz * (CCH * NPIX) + (size_t)ci * NPIX +
                           (size_t)(py * 8 + kh) * HH + (px * 8 + kw);
              tmp[j] = (bMode == 2) ? (bf16)((const float*)Bsrc)[off]
                                    : ((const bf16*)Bsrc)[off];
            }
          }
        }
      }
      stage_b_row(Bs, skk, snsg, tmp);
    }
    if (tdmA || tdmB) tdm_wait();
    __syncthreads();

    v16bf a0, a1, b0, b1;
    frag_a(As, wm * 32,      lane, a0);
    frag_a(As, wm * 32 + 16, lane, a1);
    frag_b(Bs, wn * 32,      lane, b0);
    frag_b(Bs, wn * 32 + 16, lane, b1);
    c00 = wmma_bf16(a0, b0, c00);
    c01 = wmma_bf16(a0, b1, c01);
    c10 = wmma_bf16(a1, b0, c10);
    c11 = wmma_bf16(a1, b1, c11);
    __syncthreads();
  }

  // ---- epilogue ----
  const int nOff  = lane & 15;
  const int mHalf = (lane >> 4) * 8;
#pragma unroll
  for (int j = 0; j < 8; ++j) {
    int gm0 = blkM + wm * 32 + mHalf + j;
    int gm1 = gm0 + 16;
    int gn0 = blkN + wn * 32 + nOff;
    int gn1 = gn0 + 16;
#pragma unroll
    for (int q = 0; q < 2; ++q) {
      int gm = q ? gm1 : gm0;
      if (gm >= M) continue;
      float bv = bias ? bias[gm] : 0.0f;
      float v0 = (q ? c10[j] : c00[j]) * alpha + bv;
      float v1 = (q ? c11[j] : c01[j]) * alpha + bv;
      size_t base = (size_t)bz * (size_t)dStride + (size_t)gm * N;
      if (dBf16) {
        bf16* o = (bf16*)D;
        if (gn0 < N) o[base + gn0] = (bf16)v0;
        if (gn1 < N) o[base + gn1] = (bf16)v1;
      } else {
        float* o = (float*)D;
        if (gn0 < N) o[base + gn0] = v0;
        if (gn1 < N) o[base + gn1] = v1;
      }
    }
  }
}

// ---------------------------------------------------------------------------
// 3x3 pad-1 conv as implicit GEMM (WMMA) + folded BN + ReLU. M = 256 out ch.
// mode 0: concat(f_rgb f32, f_depth f32)                 Ktot = 4608
// mode 1: concat(enh_rgb bf16, enh_depth bf16, mask f32) Ktot = 4617 (Kloop 4640)
// ---------------------------------------------------------------------------
__global__ __launch_bounds__(256) void conv3x3_wmma(
    int Kloop, int Ktot, int lda, const bf16* __restrict__ W,
    const void* __restrict__ src0, const void* __restrict__ src1,
    const float* __restrict__ maskSrc, int mode,
    const float* __restrict__ bns, const float* __restrict__ bnt,
    float* __restrict__ out) {
  __shared__ bf16 As[BM * APITCH];
  __shared__ bf16 Bs[BK * BPITCH];

  const int bz   = blockIdx.z;
  const int tid  = threadIdx.x;
  const int lane = tid & 31;
  const int wid  = tid >> 5;
  const int wm   = wid >> 1;
  const int wn   = wid & 1;
  const int blkM = blockIdx.y * BM;
  const int blkN = blockIdx.x * BN;

  const int skk  = tid >> 3;
  const int snsg = (tid & 7) << 3;
  const unsigned asOff = lds_off(As);

  v8f c00 = {}; v8f c01 = {}; v8f c10 = {}; v8f c11 = {};

  for (int k0 = 0; k0 < Kloop; k0 += BK) {
    // ---- stage A (weights, M=256: tiles always full) ----
    if (HAVE_TDM) {
      if (wid == 0) {
        unsigned long long ga =
            (unsigned long long)(size_t)(W + (size_t)blkM * lda + k0);
        tdm_load_2d(ga, asOff, BK, BM, (unsigned)(Kloop - k0), (unsigned)(CCH - blkM),
                    (unsigned long long)lda, 3, 4);
      }
    } else {
#pragma unroll
      for (int i = 0; i < 2; ++i) {
        int idx  = tid + i * 256;
        int kseg = (idx & 3) << 3;
        int mm   = idx >> 2;
        *(bf16x8*)(As + (size_t)mm * APITCH + kseg) =
            *(const bf16x8*)(W + (size_t)(blkM + mm) * lda + (k0 + kseg));
      }
    }
    // ---- stage B (implicit im2col gather) ----
    {
      int gk = k0 + skk;
      bf16 tmp[8];
#pragma unroll
      for (int j = 0; j < 8; ++j) tmp[j] = (bf16)0.0f;
      if (gk < Ktot) {
        int ci = gk / 9, r = gk - ci * 9;
        int kh = r / 3, kw = r - kh * 3;
        int gn0 = blkN + snsg;             // always < NPIX (7744 = 121*64)
        int h = gn0 / HH, w0 = gn0 - h * HH;
        int y = h + kh - 1;
        int x0 = w0 + kw - 1;
        bool fast = (w0 <= HH - 8) && (y >= 0) && (y < HH) && (x0 >= 0) && (x0 + 7 < HH);
        if (fast) {
          int sp = y * HH + x0;
          if (mode == 0) {
            const float* p = (ci < CCH)
                ? (const float*)src0 + ((size_t)bz * CCH + ci) * NPIX + sp
                : (const float*)src1 + ((size_t)bz * CCH + (ci - CCH)) * NPIX + sp;
#pragma unroll
            for (int j = 0; j < 8; ++j) tmp[j] = (bf16)p[j];
          } else if (ci < CCH) {
            const bf16* p = (const bf16*)src0 + ((size_t)bz * CCH + ci) * NPIX + sp;
#pragma unroll
            for (int j = 0; j < 8; ++j) tmp[j] = p[j];
          } else if (ci < 2 * CCH) {
            const bf16* p = (const bf16*)src1 + ((size_t)bz * CCH + (ci - CCH)) * NPIX + sp;
#pragma unroll
            for (int j = 0; j < 8; ++j) tmp[j] = p[j];
          } else {
            const float* p = maskSrc + (size_t)bz * NPIX + sp;
#pragma unroll
            for (int j = 0; j < 8; ++j) tmp[j] = (bf16)p[j];
          }
        } else {
#pragma unroll
          for (int j = 0; j < 8; ++j) {
            int gn = gn0 + j;
            int hh = gn / HH, ww = gn - hh * HH;
            int yy = hh + kh - 1, xx = ww + kw - 1;
            if (yy >= 0 && yy < HH && xx >= 0 && xx < HH) {
              int sp = yy * HH + xx;
              float v;
              if (mode == 0) {
                v = (ci < CCH)
                        ? ((const float*)src0)[((size_t)bz * CCH + ci) * NPIX + sp]
                        : ((const float*)src1)[((size_t)bz * CCH + (ci - CCH)) * NPIX + sp];
              } else if (ci < CCH) {
                v = (float)((const bf16*)src0)[((size_t)bz * CCH + ci) * NPIX + sp];
              } else if (ci < 2 * CCH) {
                v = (float)((const bf16*)src1)[((size_t)bz * CCH + (ci - CCH)) * NPIX + sp];
              } else {
                v = maskSrc[(size_t)bz * NPIX + sp];
              }
              tmp[j] = (bf16)v;
            }
          }
        }
      }
      stage_b_row(Bs, skk, snsg, tmp);
    }
    if (HAVE_TDM) tdm_wait();
    __syncthreads();

    v16bf a0, a1, b0, b1;
    frag_a(As, wm * 32,      lane, a0);
    frag_a(As, wm * 32 + 16, lane, a1);
    frag_b(Bs, wn * 32,      lane, b0);
    frag_b(Bs, wn * 32 + 16, lane, b1);
    c00 = wmma_bf16(a0, b0, c00);
    c01 = wmma_bf16(a0, b1, c01);
    c10 = wmma_bf16(a1, b0, c10);
    c11 = wmma_bf16(a1, b1, c11);
    __syncthreads();
  }

  const int nOff  = lane & 15;
  const int mHalf = (lane >> 4) * 8;
#pragma unroll
  for (int j = 0; j < 8; ++j) {
    int gm0 = blkM + wm * 32 + mHalf + j;
    int gn0 = blkN + wn * 32 + nOff;
#pragma unroll
    for (int q = 0; q < 2; ++q) {
      int gm = gm0 + q * 16;
      float s = bns[gm], t = bnt[gm];
      float v0 = fmaxf((q ? c10[j] : c00[j]) * s + t, 0.0f);
      float v1 = fmaxf((q ? c11[j] : c01[j]) * s + t, 0.0f);
      size_t base = ((size_t)bz * CCH + gm) * NPIX;
      out[base + gn0]      = v0;
      out[base + gn0 + 16] = v1;
    }
  }
}

// ---------------------------------------------------------------------------
// Small / elementwise kernels
// ---------------------------------------------------------------------------
__global__ void cvt_bf16_kernel(const float* __restrict__ s, bf16* __restrict__ d, int n) {
  int i = blockIdx.x * blockDim.x + threadIdx.x;
  if (i < n) d[i] = (bf16)s[i];
}

__global__ void cvt_pitch_kernel(const float* __restrict__ s, bf16* __restrict__ d,
                                 int K, int Kpad, int total) {
  int i = blockIdx.x * blockDim.x + threadIdx.x;
  if (i >= total) return;
  int m = i / Kpad, k = i - m * Kpad;
  d[i] = (k < K) ? (bf16)s[(size_t)m * K + k] : (bf16)0.0f;
}

__global__ void bnfold_kernel(const float* g, const float* be, const float* m,
                              const float* v, const float* cb,
                              float* s, float* t, int n) {
  int i = blockIdx.x * blockDim.x + threadIdx.x;
  if (i >= n) return;
  float sc = g[i] * rsqrtf(v[i] + LNEPS);
  s[i] = sc;
  t[i] = (cb[i] - m[i]) * sc + be[i];
}

__global__ void mask_kernel(const float* __restrict__ h1, const float* __restrict__ w2,
                            const float* __restrict__ b2, float* __restrict__ mask) {
  int i = blockIdx.x * blockDim.x + threadIdx.x;
  if (i >= NB * NPIX) return;
  int b = i / NPIX, n = i - b * NPIX;
  const float* p = h1 + (size_t)b * CCH * NPIX + n;
  float acc = b2[0];
  for (int c = 0; c < CCH; ++c) acc += p[(size_t)c * NPIX] * w2[c];
  mask[i] = 1.0f / (1.0f + __expf(-acc));
}

__global__ void maskmul_kernel(const float* __restrict__ depth,
                               const float* __restrict__ mask,
                               bf16* __restrict__ out) {
  size_t i = (size_t)blockIdx.x * blockDim.x + threadIdx.x;
  if (i >= (size_t)NB * CCH * NPIX) return;
  size_t b = i / ((size_t)CCH * NPIX);
  size_t n = i % NPIX;
  out[i] = (bf16)(depth[i] * mask[b * NPIX + n]);
}

__global__ void ln_small_kernel(const float* __restrict__ x, const float* __restrict__ g,
                                const float* __restrict__ bb, float* __restrict__ o) {
  int i = blockIdx.x * blockDim.x + threadIdx.x;
  if (i >= NB * NRED) return;
  int b = i / NRED, p = i - b * NRED;
  const float* xp = x + (size_t)b * CCH * NRED + p;
  float s = 0.f, s2 = 0.f;
  for (int c = 0; c < CCH; ++c) { float v = xp[(size_t)c * NRED]; s += v; s2 += v * v; }
  float mean = s * (1.0f / CCH);
  float var  = s2 * (1.0f / CCH) - mean * mean;
  float inv  = rsqrtf(var + LNEPS);
  float* op  = o + (size_t)b * CCH * NRED + p;
  for (int c = 0; c < CCH; ++c)
    op[(size_t)c * NRED] = (xp[(size_t)c * NRED] - mean) * inv * g[c] + bb[c];
}

__global__ void ktr_kernel(const float* __restrict__ Km, bf16* __restrict__ Kt) {
  int i = blockIdx.x * blockDim.x + threadIdx.x;
  if (i >= NB * NRED * CQ) return;
  int b = i / (NRED * CQ), r = (i / CQ) % NRED, c = i % CQ;
  Kt[i] = (bf16)Km[((size_t)b * CQ + c) * NRED + r];
}

__global__ void vpad_kernel(const float* __restrict__ V, bf16* __restrict__ Vp) {
  int i = blockIdx.x * blockDim.x + threadIdx.x;
  if (i >= NB * CCH * NREDP) return;
  int r = i % NREDP;
  int bc = i / NREDP;
  Vp[i] = (r < NRED) ? (bf16)V[(size_t)bc * NRED + r] : (bf16)0.0f;
}

__global__ void softmax_kernel(const float* __restrict__ sc, bf16* __restrict__ at) {
  int i = blockIdx.x * blockDim.x + threadIdx.x;
  if (i >= NB * NPIX) return;
  int b = i / NPIX, n = i - b * NPIX;
  const float* s = sc + (size_t)b * NRED * NPIX + n;
  float mx = -3.4e38f;
  for (int r = 0; r < NRED; ++r) mx = fmaxf(mx, s[(size_t)r * NPIX]);
  float sum = 0.f;
  for (int r = 0; r < NRED; ++r) sum += __expf(s[(size_t)r * NPIX] - mx);
  float inv = 1.0f / sum;
  bf16* o = at + (size_t)b * NREDP * NPIX + n;
  for (int r = 0; r < NRED; ++r)
    o[(size_t)r * NPIX] = (bf16)(__expf(s[(size_t)r * NPIX] - mx) * inv);
  for (int r = NRED; r < NREDP; ++r) o[(size_t)r * NPIX] = (bf16)0.0f;
}

__global__ void ln_residual_kernel(const float* __restrict__ x, const float* __restrict__ base,
                                   const float* __restrict__ g, const float* __restrict__ bb,
                                   const float* __restrict__ gamma, bf16* __restrict__ enh) {
  int i = blockIdx.x * blockDim.x + threadIdx.x;
  if (i >= NB * NPIX) return;
  int b = i / NPIX, n = i - b * NPIX;
  const float* xp = x + (size_t)b * CCH * NPIX + n;
  float s = 0.f, s2 = 0.f;
  for (int c = 0; c < CCH; ++c) { float v = xp[(size_t)c * NPIX]; s += v; s2 += v * v; }
  float mean = s * (1.0f / CCH);
  float var  = s2 * (1.0f / CCH) - mean * mean;
  float inv  = rsqrtf(var + LNEPS);
  float gc   = fminf(fmaxf(gamma[0], 0.0f), 1.0f);
  const float* bp = base + (size_t)b * CCH * NPIX + n;
  bf16* op = enh + (size_t)b * CCH * NPIX + n;
  for (int c = 0; c < CCH; ++c) {
    float v = (xp[(size_t)c * NPIX] - mean) * inv * g[c] + bb[c];
    op[(size_t)c * NPIX] = (bf16)(bp[(size_t)c * NPIX] + gc * v);
  }
}

// ---------------------------------------------------------------------------
// Host-side orchestration
// ---------------------------------------------------------------------------
static inline dim3 g1(size_t n) { return dim3((unsigned)((n + 255) / 256)); }

extern "C" void kernel_launch(void* const* d_in, const int* in_sizes, int n_in,
                              void* d_out, int out_size, void* d_ws, size_t ws_size,
                              hipStream_t stream) {
  (void)in_sizes; (void)n_in; (void)out_size; (void)ws_size;

  const float* f_rgb   = (const float*)d_in[0];
  const float* f_depth = (const float*)d_in[1];

  char* wp = (char*)d_ws;
  auto alloc = [&](size_t bytes) -> void* {
    void* p = (void*)wp;
    wp += (bytes + 255) & ~(size_t)255;
    return p;
  };

  bf16* wc1    = (bf16*)alloc((size_t)CCH * KC1 * 2);
  bf16* wfus   = (bf16*)alloc((size_t)CCH * KFUSP * 2);
  bf16* wsr[2] = { (bf16*)alloc((size_t)CCH * KSR * 2), (bf16*)alloc((size_t)CCH * KSR * 2) };
  bf16* wq[2]  = { (bf16*)alloc((size_t)CQ * CCH * 2),  (bf16*)alloc((size_t)CQ * CCH * 2) };
  bf16* wk[2]  = { (bf16*)alloc((size_t)CQ * CCH * 2),  (bf16*)alloc((size_t)CQ * CCH * 2) };
  bf16* wv[2]  = { (bf16*)alloc((size_t)CCH * CCH * 2), (bf16*)alloc((size_t)CCH * CCH * 2) };
  float* bns1 = (float*)alloc(CCH * 4);
  float* bnt1 = (float*)alloc(CCH * 4);
  float* bnsf = (float*)alloc(CCH * 4);
  float* bntf = (float*)alloc(CCH * 4);
  float* h1      = (float*)alloc((size_t)NB * CCH * NPIX * 4);
  float* mask    = (float*)alloc((size_t)NB * NPIX * 4);
  bf16*  maskedb = (bf16*) alloc((size_t)NB * CCH * NPIX * 2);
  float* reduced = (float*)alloc((size_t)NB * CCH * NRED * 4);
  float* rln     = (float*)alloc((size_t)NB * CCH * NRED * 4);
  float* Kmat    = (float*)alloc((size_t)NB * CQ * NRED * 4);
  float* Vbuf    = (float*)alloc((size_t)NB * CCH * NRED * 4);
  bf16*  Ktr     = (bf16*) alloc((size_t)NB * NRED * CQ * 2);
  bf16*  Vpad    = (bf16*) alloc((size_t)NB * CCH * NREDP * 2);
  bf16*  Qbuf    = (bf16*) alloc((size_t)NB * CQ * NPIX * 2);
  float* scoresT = (float*)alloc((size_t)NB * NRED * NPIX * 4);
  bf16*  attnT   = (bf16*) alloc((size_t)NB * NREDP * NPIX * 2);
  float* out_sra = (float*)alloc((size_t)NB * CCH * NPIX * 4);
  bf16*  enh[2]  = { (bf16*)alloc((size_t)NB * CCH * NPIX * 2),
                     (bf16*)alloc((size_t)NB * CCH * NPIX * 2) };

  auto cvt = [&](const void* s, bf16* d, size_t n) {
    cvt_bf16_kernel<<<g1(n), 256, 0, stream>>>((const float*)s, d, (int)n);
  };
  cvt(d_in[2],  wc1,    (size_t)CCH * KC1);
  cvt_pitch_kernel<<<g1((size_t)CCH * KFUSP), 256, 0, stream>>>(
      (const float*)d_in[36], wfus, KFUS, KFUSP, CCH * KFUSP);
  cvt(d_in[16], wsr[0], (size_t)CCH * KSR);
  cvt(d_in[28], wsr[1], (size_t)CCH * KSR);
  cvt(d_in[10], wq[0],  (size_t)CQ * CCH);
  cvt(d_in[22], wq[1],  (size_t)CQ * CCH);
  cvt(d_in[12], wk[0],  (size_t)CQ * CCH);
  cvt(d_in[24], wk[1],  (size_t)CQ * CCH);
  cvt(d_in[14], wv[0],  (size_t)CCH * CCH);
  cvt(d_in[26], wv[1],  (size_t)CCH * CCH);

  bnfold_kernel<<<1, 256, 0, stream>>>((const float*)d_in[4], (const float*)d_in[5],
                                       (const float*)d_in[6], (const float*)d_in[7],
                                       (const float*)d_in[3], bns1, bnt1, CCH);
  bnfold_kernel<<<1, 256, 0, stream>>>((const float*)d_in[38], (const float*)d_in[39],
                                       (const float*)d_in[40], (const float*)d_in[41],
                                       (const float*)d_in[37], bnsf, bntf, CCH);

  // conv1 (512->256, 3x3) + BN + ReLU -> h1
  conv3x3_wmma<<<dim3(121, 2, NB), 256, 0, stream>>>(
      KC1, KC1, KC1, wc1, f_rgb, f_depth, nullptr, 0, bns1, bnt1, h1);

  mask_kernel<<<g1((size_t)NB * NPIX), 256, 0, stream>>>(
      h1, (const float*)d_in[8], (const float*)d_in[9], mask);
  maskmul_kernel<<<g1((size_t)NB * CCH * NPIX), 256, 0, stream>>>(f_depth, mask, maskedb);

  for (int s = 0; s < 2; ++s) {
    int ib = 10 + 12 * s;
    const float* q_b    = (const float*)d_in[ib + 1];
    const float* k_b    = (const float*)d_in[ib + 3];
    const float* v_b    = (const float*)d_in[ib + 5];
    const float* sr_b   = (const float*)d_in[ib + 7];
    const float* ln_g   = (const float*)d_in[ib + 8];
    const float* ln_b   = (const float*)d_in[ib + 9];
    const float* norm_g = (const float*)d_in[ib + 10];
    const float* norm_b = (const float*)d_in[ib + 11];
    const float* gamma  = (const float*)d_in[34 + s];

    const void* kvSrc = (s == 0) ? (const void*)maskedb : (const void*)f_rgb;
    int kvMode        = (s == 0) ? 3 : 2;
    const void* qSrc  = (s == 0) ? (const void*)f_rgb : (const void*)maskedb;
    int qMode         = (s == 0) ? 0 : 1;
    const float* baseRes = (s == 0) ? f_rgb : f_depth;

    // sr conv (8x8 stride 8): [256 x 16384] x gather -> reduced f32 [256x121]
    gemm_wmma<<<dim3(2, 2, NB), 256, 0, stream>>>(
        CCH, KSR, NRED, wsr[s], KSR, 0,
        kvSrc, kvMode, 0, 0,
        reduced, (long long)CCH * NRED, 0, sr_b, 1.0f);

    ln_small_kernel<<<g1((size_t)NB * NRED), 256, 0, stream>>>(reduced, ln_g, ln_b, rln);

    gemm_wmma<<<dim3(2, 1, NB), 256, 0, stream>>>(
        CQ, CCH, NRED, wk[s], CCH, 0,
        rln, 0, NRED, (long long)CCH * NRED,
        Kmat, (long long)CQ * NRED, 0, k_b, 1.0f);
    gemm_wmma<<<dim3(2, 2, NB), 256, 0, stream>>>(
        CCH, CCH, NRED, wv[s], CCH, 0,
        rln, 0, NRED, (long long)CCH * NRED,
        Vbuf, (long long)CCH * NRED, 0, v_b, 1.0f);

    ktr_kernel<<<g1((size_t)NB * NRED * CQ), 256, 0, stream>>>(Kmat, Ktr);
    vpad_kernel<<<g1((size_t)NB * CCH * NREDP), 256, 0, stream>>>(Vbuf, Vpad);

    // Q: [32x256] x qSrc -> Qbuf bf16 [32x7744]
    gemm_wmma<<<dim3(121, 1, NB), 256, 0, stream>>>(
        CQ, CCH, NPIX, wq[s], CCH, 0,
        qSrc, qMode, NPIX, (long long)CCH * NPIX,
        Qbuf, (long long)CQ * NPIX, 1, q_b, 1.0f);

    // scores^T = K^T Q * scale : A = Ktr [121x32] per batch, B = Qbuf bf16
    gemm_wmma<<<dim3(121, 1, NB), 256, 0, stream>>>(
        NRED, CQ, NPIX, Ktr, CQ, (long long)NRED * CQ,
        Qbuf, 1, NPIX, (long long)CQ * NPIX,
        scoresT, (long long)NRED * NPIX, 0, nullptr, 0.17677669529663687f);

    softmax_kernel<<<g1((size_t)NB * NPIX), 256, 0, stream>>>(scoresT, attnT);

    // out = V * attn^T : A = Vpad [256x128] per batch, B = attnT bf16 [128x7744]
    gemm_wmma<<<dim3(121, 2, NB), 256, 0, stream>>>(
        CCH, NREDP, NPIX, Vpad, NREDP, (long long)CCH * NREDP,
        attnT, 1, NPIX, (long long)NREDP * NPIX,
        out_sra, (long long)CCH * NPIX, 0, nullptr, 1.0f);

    ln_residual_kernel<<<g1((size_t)NB * NPIX), 256, 0, stream>>>(
        out_sra, baseRes, norm_g, norm_b, gamma, enh[s]);
  }

  // fusion conv (513->256, 3x3, padded K) + BN + ReLU -> d_out
  conv3x3_wmma<<<dim3(121, 2, NB), 256, 0, stream>>>(
      KFUSP, KFUS, KFUSP, wfus, enh[0], enh[1], mask, 1, bnsf, bntf, (float*)d_out);
}
